// CausalSelfAttention_12678743458150
// MI455X (gfx1250) — compile-verified, matrix operand order checked by reading, not run on hardware
//
#include <hip/hip_runtime.h>
#include <hip/hip_bf16.h>

// ---------------------------------------------------------------------------
// CausalSelfAttention for MI455X (gfx1250, wave32, WMMA).
// All GEMMs run on v_wmma_f32_16x16x32_bf16 (bf16 in, fp32 accumulate).
// GEMM tiles are staged with GLOBAL_LOAD_ASYNC_TO_LDS (ASYNCcnt) and
// double-buffered LDS; softmax / RMSNorm / RoPE stay in fp32 VALU.
//
// Workspace layout (d_ws), ~120 MB total (see launcher).
// ---------------------------------------------------------------------------

typedef __attribute__((ext_vector_type(16))) __bf16 v16bf;
typedef __attribute__((ext_vector_type(8)))  float  v8f;

#if __has_builtin(__builtin_amdgcn_global_load_async_to_lds_b128)
#define HAVE_ASYNC_LDS 1
typedef int v4i_vec __attribute__((vector_size(16)));
typedef __attribute__((address_space(1))) v4i_vec glb_b128;   // global int4
typedef __attribute__((address_space(3))) v4i_vec lds_b128;   // LDS int4
#else
#define HAVE_ASYNC_LDS 0
#endif

#if __has_builtin(__builtin_amdgcn_s_wait_asynccnt)
#define WAIT_ASYNC(n) __builtin_amdgcn_s_wait_asynccnt(n)
#else
#define WAIT_ASYNC(n) asm volatile("s_wait_asynccnt %0" ::"i"(n) : "memory")
#endif

union FragU {
  uint4 q[2];   // 32 bytes
  v16bf v;      // 16 x bf16
};

static __device__ __forceinline__ unsigned short f2bf(float f) {
  unsigned int u = __builtin_bit_cast(unsigned int, f);
  u += 0x7FFFu + ((u >> 16) & 1u);       // round-to-nearest-even
  return (unsigned short)(u >> 16);
}
static __device__ __forceinline__ float bf2f(unsigned short h) {
  unsigned int u = ((unsigned int)h) << 16;
  return __builtin_bit_cast(float, u);
}

// ---------------------------- fp32 -> bf16 convert -------------------------
__global__ void __launch_bounds__(256)
cvt_f32_bf16(const float* __restrict__ in, unsigned short* __restrict__ out, int n4) {
  int i = blockIdx.x * 256 + threadIdx.x;
  if (i < n4) {
    float4 f = ((const float4*)in)[i];
    unsigned long long p =
        (unsigned long long)f2bf(f.x)
      | ((unsigned long long)f2bf(f.y) << 16)
      | ((unsigned long long)f2bf(f.z) << 32)
      | ((unsigned long long)f2bf(f.w) << 48);
    ((unsigned long long*)out)[i] = p;
  }
}

// ---------------------------- tiled bf16 GEMM ------------------------------
// C[M,N] (+fp32 acc) = A[MxK] * B[NxK]^T, A/B bf16 row-major.
// Block tile 128x128, 8 waves (4x2), each wave 32x64 = 2x4 WMMA tiles.
// Double-buffered LDS, async global->LDS staging (4 async insts per wave/tile).

static __device__ __forceinline__ void stage_tile(
    unsigned short* lA, unsigned short* lB,
    const unsigned short* __restrict__ gA, const unsigned short* __restrict__ gB,
    int tid, int K, int m0, int n0, int k0) {
  #pragma unroll
  for (int p = 0; p < 2; ++p) {
    int idx = tid + p * 256;                 // 0..511
    int row = idx >> 2;
    int c16 = idx & 3;
    const unsigned short* sa = gA + (size_t)(m0 + row) * K + k0 + c16 * 8;
    const unsigned short* sb = gB + (size_t)(n0 + row) * K + k0 + c16 * 8;
#if HAVE_ASYNC_LDS
    __builtin_amdgcn_global_load_async_to_lds_b128(
        (glb_b128*)sa, (lds_b128*)(lA + idx * 8), 0, 0);
    __builtin_amdgcn_global_load_async_to_lds_b128(
        (glb_b128*)sb, (lds_b128*)(lB + idx * 8), 0, 0);
#else
    ((uint4*)lA)[idx] = *(const uint4*)sa;
    ((uint4*)lB)[idx] = *(const uint4*)sb;
#endif
  }
}

template<bool OUT_BF16>
__global__ void __launch_bounds__(256)
gemm_bf16_wmma(const unsigned short* __restrict__ A,
               const unsigned short* __restrict__ B,
               void* __restrict__ C, int M, int N, int K) {
  __shared__ unsigned short As[2][128 * 32];
  __shared__ unsigned short Bs[2][128 * 32];

  const int tid  = threadIdx.x;
  const int lane = tid & 31;
  const int wid  = tid >> 5;
  const int l16  = lane & 15;
  const int lh   = (lane >> 4) & 1;
  const int b0   = lh ? 8 : 0;          // K-half select within a fragment
  const int n0   = blockIdx.x * 128;
  const int m0   = blockIdx.y * 128;
  const int wm   = wid >> 1;            // 0..3 -> 32 rows each
  const int wn   = wid & 1;             // 0..1 -> 64 cols each

  v8f acc[2][4];
  for (int i = 0; i < 2; ++i)
    for (int j = 0; j < 4; ++j)
      for (int r = 0; r < 8; ++r) acc[i][j][r] = 0.0f;

  const int nk = K >> 5;
  stage_tile(As[0], Bs[0], A, B, tid, K, m0, n0, 0);

  for (int ch = 0; ch < nk; ++ch) {
    const int cur = ch & 1;
    if (ch + 1 < nk) {
      stage_tile(As[cur ^ 1], Bs[cur ^ 1], A, B, tid, K, m0, n0, (ch + 1) << 5);
#if HAVE_ASYNC_LDS
      WAIT_ASYNC(4);                    // current tile resident, next in flight
#endif
    } else {
#if HAVE_ASYNC_LDS
      WAIT_ASYNC(0);
#endif
    }
    __syncthreads();

    FragU af[2], bf[4];
    for (int i = 0; i < 2; ++i) {
      int r = wm * 32 + i * 16 + l16;
      af[i].q[0] = *(const uint4*)(As[cur] + r * 32 + b0);
      af[i].q[1] = *(const uint4*)(As[cur] + r * 32 + 16 + b0);
    }
    for (int j = 0; j < 4; ++j) {
      int r = wn * 64 + j * 16 + l16;
      bf[j].q[0] = *(const uint4*)(Bs[cur] + r * 32 + b0);
      bf[j].q[1] = *(const uint4*)(Bs[cur] + r * 32 + 16 + b0);
    }
    #pragma unroll
    for (int i = 0; i < 2; ++i)
      #pragma unroll
      for (int j = 0; j < 4; ++j)
        acc[i][j] = __builtin_amdgcn_wmma_f32_16x16x32_bf16(
            false, af[i].v, false, bf[j].v, (short)0, acc[i][j], false, false);
    __syncthreads();
  }

  for (int i = 0; i < 2; ++i)
    for (int j = 0; j < 4; ++j)
      for (int r = 0; r < 8; ++r) {
        int row = m0 + wm * 32 + i * 16 + r + 8 * lh;
        int col = n0 + wn * 64 + j * 16 + l16;
        float v = acc[i][j][r];
        if (OUT_BF16) ((unsigned short*)C)[(size_t)row * N + col] = f2bf(v);
        else          ((float*)C)[(size_t)row * N + col] = v;
      }
}

// ------------------------- RoPE + RMSNorm + scatter ------------------------
// One wave per (b, t, slot); slots: 0..15 q heads, 16..23 k heads, 24..31 v.
// Lane owns dims {lane, lane+32, lane+64, lane+96} of the 128-wide head.
__global__ void __launch_bounds__(256)
rope_rms_scatter(const unsigned short* __restrict__ qkv,   // [4096][4096] bf16
                 const float* __restrict__ cosT,           // [2048][64]
                 const float* __restrict__ sinT,
                 unsigned short* __restrict__ qb,          // [B*16][2048][128]
                 unsigned short* __restrict__ kbuf,        // [B*8][2048][128]
                 unsigned short* __restrict__ vbT) {       // [B*8][128][2048]
  const int lane = threadIdx.x & 31;
  int s    = blockIdx.x * 8 + (threadIdx.x >> 5);   // 0 .. 131071
  int b    = s >> 16;                                // / (2048*32)
  int rem  = s & 65535;
  int t    = rem >> 5;
  int slot = rem & 31;

  const unsigned short* src = qkv + (size_t)(b * 2048 + t) * 4096;

  if (slot >= 24) {                                   // v: transpose copy only
    int h = slot - 24;
    const unsigned short* vsrc = src + 3072 + h * 128;
    size_t base = ((size_t)(b * 8 + h) * 128) * 2048 + t;
    #pragma unroll
    for (int p = 0; p < 4; ++p) {
      int d = lane + p * 32;
      vbT[base + (size_t)d * 2048] = vsrc[d];
    }
    return;
  }

  bool isq = slot < 16;
  int  h   = isq ? slot : (slot - 16);
  const unsigned short* hsrc = src + (isq ? h * 128 : 2048 + h * 128);

  float d0 = bf2f(hsrc[lane]);
  float d1 = bf2f(hsrc[lane + 32]);
  float d2 = bf2f(hsrc[lane + 64]);
  float d3 = bf2f(hsrc[lane + 96]);
  float cs0 = cosT[t * 64 + lane],      sn0 = sinT[t * 64 + lane];
  float cs1 = cosT[t * 64 + lane + 32], sn1 = sinT[t * 64 + lane + 32];

  float y0 =  d0 * cs0 + d2 * sn0;   // pair (lane, lane+64)
  float y2 = -d0 * sn0 + d2 * cs0;
  float y1 =  d1 * cs1 + d3 * sn1;   // pair (lane+32, lane+96)
  float y3 = -d1 * sn1 + d3 * cs1;

  float ss = y0 * y0 + y1 * y1 + y2 * y2 + y3 * y3;
  for (int off = 1; off < 32; off <<= 1) ss += __shfl_xor(ss, off, 32);
  float sc = rsqrtf(ss * (1.0f / 128.0f) + 1.1920929e-07f);
  if (isq) sc *= 0.08838834764831845f;   // fold 1/sqrt(HEAD_DIM) into q

  unsigned short* dst = isq ? (qb + ((size_t)(b * 16 + h) * 2048 + t) * 128)
                            : (kbuf + ((size_t)(b * 8 + h) * 2048 + t) * 128);
  dst[lane]      = f2bf(y0 * sc);
  dst[lane + 32] = f2bf(y1 * sc);
  dst[lane + 64] = f2bf(y2 * sc);
  dst[lane + 96] = f2bf(y3 * sc);
}

// ------------------------------ flash attention ----------------------------
// One wave per 16-query tile; online softmax over 32-key chunks.
__global__ void __launch_bounds__(256)
attn_wmma(const unsigned short* __restrict__ qb,
          const unsigned short* __restrict__ kbuf,
          const unsigned short* __restrict__ vbT,
          unsigned short* __restrict__ yb) {          // [4096][2048] bf16
  __shared__ unsigned short Ps[8][16 * 32];           // per-wave P bounce

  const int lane = threadIdx.x & 31;
  const int w    = threadIdx.x >> 5;
  const int l16  = lane & 15;
  const int lh   = (lane >> 4) & 1;
  const int kb0  = lh ? 8 : 0;

  int tile = blockIdx.x * 8 + w;                      // 0..4095
  int q0   = (tile & 127) << 4;
  int bh   = tile >> 7;
  int b    = bh >> 4;
  int h    = bh & 15;
  int kvh  = h >> 1;                                  // GQA: repeat(k,2)

  const unsigned short* Q  = qb  + ((size_t)(b * 16 + h) * 2048 + q0) * 128;
  const unsigned short* Kp = kbuf + (size_t)(b * 8 + kvh) * 2048 * 128;
  const unsigned short* Vp = vbT  + (size_t)(b * 8 + kvh) * 128 * 2048;

  FragU qf[4];
  {
    const unsigned short* qrow = Q + (size_t)l16 * 128;
    #pragma unroll
    for (int kd = 0; kd < 4; ++kd) {
      qf[kd].q[0] = *(const uint4*)(qrow + kd * 32 + kb0);
      qf[kd].q[1] = *(const uint4*)(qrow + kd * 32 + 16 + kb0);
    }
  }

  v8f O[8];
  float mrow[8], lrow[8];
  for (int dj = 0; dj < 8; ++dj)
    for (int r = 0; r < 8; ++r) O[dj][r] = 0.0f;
  for (int r = 0; r < 8; ++r) { mrow[r] = -1e30f; lrow[r] = 0.0f; }

  int nch = ((q0 + 15) >> 5) + 1;
  for (int ch = 0; ch < nch; ++ch) {
    int c0 = ch << 5;

    if (ch + 1 < nch) {                 // warm next key/value chunk (L0/L2)
      __builtin_prefetch(Kp + (size_t)(c0 + 32 + lane) * 128, 0, 0);
      __builtin_prefetch(Vp + (size_t)(lane * 4) * 2048 + c0 + 32, 0, 0);
    }

    // S = Q * K^T for 32 keys (two 16-key N tiles), contraction over d=128
    v8f S0, S1;
    for (int r = 0; r < 8; ++r) { S0[r] = 0.0f; S1[r] = 0.0f; }
    #pragma unroll
    for (int kd = 0; kd < 4; ++kd) {
      FragU k0f, k1f;
      const unsigned short* kr0 = Kp + (size_t)(c0 + l16) * 128 + kd * 32;
      const unsigned short* kr1 = Kp + (size_t)(c0 + 16 + l16) * 128 + kd * 32;
      k0f.q[0] = *(const uint4*)(kr0 + kb0);
      k0f.q[1] = *(const uint4*)(kr0 + 16 + kb0);
      k1f.q[0] = *(const uint4*)(kr1 + kb0);
      k1f.q[1] = *(const uint4*)(kr1 + 16 + kb0);
      S0 = __builtin_amdgcn_wmma_f32_16x16x32_bf16(false, qf[kd].v, false, k0f.v,
                                                   (short)0, S0, false, false);
      S1 = __builtin_amdgcn_wmma_f32_16x16x32_bf16(false, qf[kd].v, false, k1f.v,
                                                   (short)0, S1, false, false);
    }

    bool needmask = (c0 + 31 > q0);
    #pragma unroll
    for (int r = 0; r < 8; ++r) {
      float s0 = S0[r], s1 = S1[r];
      if (needmask) {
        int row = q0 + r + 8 * lh;
        if (c0 + l16 > row)      s0 = -1e30f;
        if (c0 + 16 + l16 > row) s1 = -1e30f;
      }
      float mv = fmaxf(s0, s1);
      for (int off = 1; off < 16; off <<= 1) mv = fmaxf(mv, __shfl_xor(mv, off, 32));
      float mn = fmaxf(mrow[r], mv);
      float al = __expf(mrow[r] - mn);
      mrow[r] = mn;
      float p0 = __expf(s0 - mn);
      float p1 = __expf(s1 - mn);
      S0[r] = p0; S1[r] = p1;
      float rs = p0 + p1;
      for (int off = 1; off < 16; off <<= 1) rs += __shfl_xor(rs, off, 32);
      lrow[r] = lrow[r] * al + rs;
      #pragma unroll
      for (int dj = 0; dj < 8; ++dj) O[dj][r] *= al;
    }

    // Transpose P (C layout) -> A-fragment layout via per-wave LDS bounce.
    unsigned short* ps = Ps[w];
    #pragma unroll
    for (int r = 0; r < 8; ++r) {
      int prow = r + 8 * lh;
      ps[prow * 32 + l16]      = f2bf(S0[r]);
      ps[prow * 32 + 16 + l16] = f2bf(S1[r]);
    }
    asm volatile("s_wait_dscnt 0" ::: "memory");   // same-wave store->load
    FragU pf;
    pf.q[0] = *(const uint4*)(ps + l16 * 32 + kb0);
    pf.q[1] = *(const uint4*)(ps + l16 * 32 + 16 + kb0);

    // O += P * V ; V^T stored [d][t] so key-dim fragments are contiguous b128
    #pragma unroll
    for (int dj = 0; dj < 8; ++dj) {
      FragU vf;
      const unsigned short* vr = Vp + (size_t)(dj * 16 + l16) * 2048 + c0;
      vf.q[0] = *(const uint4*)(vr + kb0);
      vf.q[1] = *(const uint4*)(vr + 16 + kb0);
      O[dj] = __builtin_amdgcn_wmma_f32_16x16x32_bf16(false, pf.v, false, vf.v,
                                                      (short)0, O[dj], false, false);
    }
  }

  #pragma unroll
  for (int r = 0; r < 8; ++r) {
    float inv = 1.0f / lrow[r];
    int trow = q0 + r + 8 * lh;
    size_t base = ((size_t)(b * 2048 + trow)) * 2048 + h * 128;
    for (int dj = 0; dj < 8; ++dj)
      yb[base + dj * 16 + l16] = f2bf(O[dj][r] * inv);
  }
}

// ------------------------------- launcher ----------------------------------
extern "C" void kernel_launch(void* const* d_in, const int* in_sizes, int n_in,
                              void* d_out, int out_size, void* d_ws, size_t ws_size,
                              hipStream_t stream) {
  const float* x      = (const float*)d_in[0];   // [2,2048,2048]
  const float* w_attn = (const float*)d_in[1];   // [4096,2048]
  const float* w_proj = (const float*)d_in[2];   // [2048,2048]
  const float* cosT   = (const float*)d_in[3];   // [2048,64]
  const float* sinT   = (const float*)d_in[4];

  char* p = (char*)d_ws;
  unsigned short* xb   = (unsigned short*)p; p += 16777216;  // 4096x2048 bf16
  unsigned short* wat  = (unsigned short*)p; p += 16777216;  // 4096x2048 bf16
  unsigned short* wpr  = (unsigned short*)p; p +=  8388608;  // 2048x2048 bf16
  unsigned short* qkvb = (unsigned short*)p; p += 33554432;  // 4096x4096 bf16
  unsigned short* qb   = (unsigned short*)p; p += 16777216;
  unsigned short* kbuf = (unsigned short*)p; p +=  8388608;
  unsigned short* vbT  = (unsigned short*)p; p +=  8388608;
  unsigned short* yb   = (unsigned short*)p; p += 16777216;  // ~120 MB total

  // 1) fp32 -> bf16 conversions
  cvt_f32_bf16<<<8192, 256, 0, stream>>>(x,      xb,  2097152);
  cvt_f32_bf16<<<8192, 256, 0, stream>>>(w_attn, wat, 2097152);
  cvt_f32_bf16<<<4096, 256, 0, stream>>>(w_proj, wpr, 1048576);

  // 2) qkv = x @ w_attn^T   (M=4096, N=4096, K=2048) -> bf16
  gemm_bf16_wmma<true><<<dim3(32, 32), 256, 0, stream>>>(xb, wat, qkvb, 4096, 4096, 2048);

  // 3) RoPE + RMSNorm + head scatter (v transposed)
  rope_rms_scatter<<<16384, 256, 0, stream>>>(qkvb, cosT, sinT, qb, kbuf, vbT);

  // 4) causal flash attention -> yb (bf16)
  attn_wmma<<<512, 256, 0, stream>>>(qb, kbuf, vbT, yb);

  // 5) out = y @ w_proj^T   (M=4096, N=2048, K=2048) -> fp32 d_out
  gemm_bf16_wmma<false><<<dim3(16, 32), 256, 0, stream>>>(yb, wpr, d_out, 4096, 2048, 2048);
}
